// OPTAttention_Mask_48129403519466
// MI455X (gfx1250) — matrix operand order repaired and checked by reading, not compiled, not run on hardware
//
#include <hip/hip_runtime.h>

#define DEV __device__ __forceinline__

typedef __bf16 bf16_t;
typedef __attribute__((ext_vector_type(16))) __bf16 v16bf;
typedef __attribute__((ext_vector_type(8)))  float  v8f;

namespace {
constexpr int   Bb      = 2, T = 2048, E = 1024, H = 16, D = 64, BH = Bb * H;
constexpr int   HB      = 204;            // int(0.1 * 2048)
constexpr int   RB      = 204;
constexpr float PENALTY = 0.98f;
constexpr float QSCALE  = 0.125f;         // D^-0.5
constexpr float FNEG    = -3.402823466e38f;   // finfo(f32).min
constexpr float MNEG    = -1.0e9f;        // min(attention_mask)
}

// ---------------- helpers ----------------

DEV bf16_t f2bf(float x) {
  unsigned u = __builtin_bit_cast(unsigned, x);
  unsigned r = u + 0x7FFFu + ((u >> 16) & 1u);
  unsigned short h = (unsigned short)(r >> 16);
  return __builtin_bit_cast(bf16_t, h);
}

template <typename T_> DEV T_ cvt_out(float x);
template <> DEV float  cvt_out<float>(float x)  { return x; }
template <> DEV bf16_t cvt_out<bf16_t>(float x) { return f2bf(x); }

// Async global->LDS 128b copy (one wave-level instruction; 32 lanes x 16B).
// lds: byte offset within the block's LDS allocation (low 32 bits of the
// generic shared-pointer, per the flat->LDS truncation rule).
DEV void async_copy_b128(unsigned lds, const void* gptr) {
  unsigned long long ga = (unsigned long long)(uintptr_t)gptr;
  asm volatile("global_load_async_to_lds_b128 %0, %1, off"
               :: "v"(lds), "v"(ga) : "memory");
}

DEV void wait_async0() {
#if __has_builtin(__builtin_amdgcn_s_wait_asynccnt)
  __builtin_amdgcn_s_wait_asynccnt(0);
#else
  asm volatile("s_wait_asynccnt 0" ::: "memory");
#endif
}

// A fragment, 16-bit 16x32 (M x K).  Lanes 0-15: M=lane, K 0..15;
// lanes 16-31: M=lane-16, K 16..31. Each VGPR packs K, K+1.
DEV v16bf load_a(const bf16_t* A, int lda, int mBase, int kBase, int lane) {
  int m = mBase + (lane & 15);
  int kg = (lane < 16) ? 0 : 16;
  const bf16_t* row = A + (size_t)m * lda + kBase;
  union { v16bf v; unsigned u[8]; } r;
#pragma unroll
  for (int i = 0; i < 8; ++i) {
    int k0 = 2 * i + kg;
    r.u[i] = *(const unsigned*)(row + k0);
  }
  return r.v;
}

// B fragment (32x16 K x N) where B[k][n] = Bt[n*ldb + k]  (weight W[N,K] / K-matrix).
DEV v16bf load_b_T(const bf16_t* Bt, int ldb, int nBase, int kBase, int lane) {
  int n = nBase + (lane & 15);
  int kg = (lane < 16) ? 0 : 16;
  const bf16_t* row = Bt + (size_t)n * ldb + kBase + kg;
  union { v16bf v; unsigned u[8]; } r;
#pragma unroll
  for (int i = 0; i < 8; ++i) r.u[i] = *(const unsigned*)(row + 2 * i);
  return r.v;
}

// B fragment where B is row-major [K,N]: B[k][n] = Bm[k*ldb + n]  (V matrix).
DEV v16bf load_b_N(const bf16_t* Bm, int ldb, int nBase, int kBase, int lane) {
  int n = nBase + (lane & 15);
  int kg = (lane < 16) ? 0 : 16;
  union { v16bf v; bf16_t h[16]; } r;
#pragma unroll
  for (int i = 0; i < 8; ++i) {
    int k = kBase + kg + 2 * i;
    r.h[2 * i]     = Bm[(size_t)k * ldb + n];
    r.h[2 * i + 1] = Bm[(size_t)(k + 1) * ldb + n];
  }
  return r.v;
}

DEV float warpMax(float v) {
#pragma unroll
  for (int o = 16; o > 0; o >>= 1) v = fmaxf(v, __shfl_xor(v, o, 32));
  return v;
}
DEV float warpSum(float v) {
#pragma unroll
  for (int o = 16; o > 0; o >>= 1) v += __shfl_xor(v, o, 32);
  return v;
}

DEV float blockMax(float v) {
  __shared__ float s[32];
  int wave = threadIdx.x >> 5, lane = threadIdx.x & 31;
  v = warpMax(v);
  if (lane == 0) s[wave] = v;
  __syncthreads();
  if (wave == 0) {
    float t = (lane < (int)(blockDim.x >> 5)) ? s[lane] : FNEG;
    t = warpMax(t);
    if (lane == 0) s[0] = t;
  }
  __syncthreads();
  float r = s[0];
  __syncthreads();
  return r;
}

DEV float blockSum(float v) {
  __shared__ float s[32];
  int wave = threadIdx.x >> 5, lane = threadIdx.x & 31;
  v = warpSum(v);
  if (lane == 0) s[wave] = v;
  __syncthreads();
  if (wave == 0) {
    float t = (lane < (int)(blockDim.x >> 5)) ? s[lane] : 0.f;
    t = warpSum(t);
    if (lane == 0) s[0] = t;
  }
  __syncthreads();
  float r = s[0];
  __syncthreads();
  return r;
}

// ---------------- kernels ----------------

__global__ void k_f32_to_bf16(const float* __restrict__ in, bf16_t* __restrict__ out, int n) {
  int i = blockIdx.x * blockDim.x + threadIdx.x;
  if (i < n) out[i] = f2bf(in[i]);
}

// LDS-staged GEMM:  Out[m,n] = (sum_k A[m,k]*W[n,k] + bias[n]) * scale.
// Block tile 64x64, BK=32; 8 waves in a 4(M) x 2(N) grid, each wave 16x32
// (two accumulators sharing one A fragment).  Operands staged via
// global_load_async_to_lds_b128 (512B per wave per tile).
template <typename OutT>
__global__ void __launch_bounds__(256)
k_gemm_bias(const bf16_t* __restrict__ A, const bf16_t* __restrict__ W,
            const float* __restrict__ bias, float scale,
            OutT* __restrict__ Out, int lda, int ldw, int ldo,
            int K, int tilesN) {
  __shared__ bf16_t As[64 * 32];   // [m][k]   4 KB
  __shared__ bf16_t Bs[64 * 32];   // [n][k]   4 KB
  int lane = threadIdx.x & 31, wave = threadIdx.x >> 5;
  int mBase = (blockIdx.x / tilesN) * 64, nBase = (blockIdx.x % tilesN) * 64;
  int waveM = (wave & 3) * 16;     // 0,16,32,48
  int waveN = (wave >> 2) * 32;    // 0,32
  v8f c0 = {0.f, 0.f, 0.f, 0.f, 0.f, 0.f, 0.f, 0.f};
  v8f c1 = {0.f, 0.f, 0.f, 0.f, 0.f, 0.f, 0.f, 0.f};

  int t = threadIdx.x;
  int rowC = t >> 2, partC = t & 3;                 // 256 chunks of 16B over 64x32
  unsigned ldsA = ((unsigned)(uintptr_t)As) + rowC * 64 + partC * 16;
  unsigned ldsB = ((unsigned)(uintptr_t)Bs) + rowC * 64 + partC * 16;
  const bf16_t* gA = A + (size_t)(mBase + rowC) * lda + partC * 8;
  const bf16_t* gB = W + (size_t)(nBase + rowC) * ldw + partC * 8;

  for (int k = 0; k < K; k += 32) {
    async_copy_b128(ldsA, gA + k);
    async_copy_b128(ldsB, gB + k);
    wait_async0();
    __syncthreads();
    v16bf a  = load_a(As, 32, waveM, 0, lane);
    v16bf b0 = load_b_T(Bs, 32, waveN, 0, lane);
    v16bf b1 = load_b_T(Bs, 32, waveN + 16, 0, lane);
    c0 = __builtin_amdgcn_wmma_f32_16x16x32_bf16(false, a, false, b0, (short)0, c0, false, false);
    c1 = __builtin_amdgcn_wmma_f32_16x16x32_bf16(false, a, false, b1, (short)0, c1, false, false);
    __syncthreads();
  }
  int n0 = nBase + waveN + (lane & 15), n1 = n0 + 16;
  float bv0 = bias[n0], bv1 = bias[n1];
  int mo = (lane >> 4) * 8;
#pragma unroll
  for (int r = 0; r < 8; ++r) {
    int m = mBase + waveM + mo + r;
    Out[(size_t)m * ldo + n0] = cvt_out<OutT>((c0[r] + bv0) * scale);
    Out[(size_t)m * ldo + n1] = cvt_out<OutT>((c1[r] + bv1) * scale);
  }
}

// aw[bh,m,n] = max(q_bh[m,:] . k_bh[n,:] + amask[b,m,n], FNEG)   (q already scaled)
// K = D = 64 -> only two WMMA steps; direct global b128 loads are already optimal.
__global__ void __launch_bounds__(256)
k_scores(const bf16_t* __restrict__ Qb, const bf16_t* __restrict__ Kb,
         const float* __restrict__ amask, float* __restrict__ aw) {
  int bh = blockIdx.y, b = bh >> 4, h = bh & 15;
  int lane = threadIdx.x & 31, wave = threadIdx.x >> 5;
  int tile = blockIdx.x * 8 + wave;          // 128x128 tiles of 16x16
  int mBase = (tile >> 7) * 16, nBase = (tile & 127) * 16;
  const bf16_t* Aq = Qb + (size_t)b * T * E + h * D;   // lda = E
  const bf16_t* Bt = Kb + (size_t)b * T * E + h * D;   // ldb = E
  v8f c = {0.f, 0.f, 0.f, 0.f, 0.f, 0.f, 0.f, 0.f};
  for (int k = 0; k < D; k += 32) {
    v16bf a = load_a(Aq, E, mBase, k, lane);
    v16bf bfr = load_b_T(Bt, E, nBase, k, lane);
    c = __builtin_amdgcn_wmma_f32_16x16x32_bf16(false, a, false, bfr, (short)0, c, false, false);
  }
  int n = nBase + (lane & 15);
  int mo = (lane >> 4) * 8;
#pragma unroll
  for (int r = 0; r < 8; ++r) {
    int m = mBase + mo + r;
    float v = c[r] + amask[((size_t)b * T + m) * T + n];
    aw[((size_t)bh * T + m) * T + n] = fmaxf(v, FNEG);
  }
}

// Sequential H2O scan per (b,h) row.  1024 threads; each owns cols tid and tid+1024.
// top-(HB-1) via 22-step threshold bisection on the non-negative accumulator.
__global__ void __launch_bounds__(1024)
k_h2o(const float* __restrict__ aw, unsigned char* __restrict__ msk) {
  int bh = blockIdx.x, tid = threadIdx.x;
  const float* A = aw + (size_t)bh * T * T;
  unsigned char* M = msk + (size_t)bh * T * T;
  int j0 = tid, j1 = tid + 1024;
  float acc0 = 0.f, acc1 = 0.f;
  // phase 1: accumulate penalty-weighted softmax of the first HB rows
  for (int r = 0; r < HB; ++r) {
    const float* row = A + (size_t)r * T;
    float v0 = row[j0], v1 = row[j1];
    float mx = blockMax(fmaxf(v0, v1));
    float e0 = __expf(v0 - mx), e1 = __expf(v1 - mx);
    float s = blockSum(e0 + e1);
    float w = powf(PENALTY, (float)(HB - 1 - r)) / s;
    acc0 += e0 * w;
    acc1 += e1 * w;
  }
  bool p0 = (j0 < HB), p1 = (j1 < HB);
  if (!p0) acc0 = 0.f;
  if (!p1) acc1 = 0.f;
  // phase 2: token-by-token scan
  for (int ti = HB; ti < T; ++ti) {
    const float* row = A + (size_t)ti * T;
    float v0 = p0 ? row[j0] : FNEG;
    float v1 = p1 ? row[j1] : FNEG;
    float mx = blockMax(fmaxf(v0, v1));
    float e0 = p0 ? __expf(v0 - mx) : 0.f;
    float e1 = p1 ? __expf(v1 - mx) : 0.f;
    float s = blockSum(e0 + e1);
    float inv = 1.f / s;
    acc0 = acc0 * PENALTY + e0 * inv;
    acc1 = acc1 * PENALTY + e1 * inv;
    // threshold for top-(HB-1)
    float hi = blockMax(fmaxf(acc0, acc1));
    float lo = 0.f;
    for (int it = 0; it < 22; ++it) {
      float thr = 0.5f * (lo + hi);
      float cnt = blockSum((acc0 > thr ? 1.f : 0.f) + (acc1 > thr ? 1.f : 0.f));
      if (cnt > (float)(HB - 1)) lo = thr; else hi = thr;
    }
    bool m0 = (acc0 > hi) || (j0 == ti);
    bool m1 = (acc1 > hi) || (j1 == ti);
    if (!m0) acc0 = 0.f;
    if (!m1) acc1 = 0.f;
    p0 = m0; p1 = m1;
    unsigned char* mr = M + (size_t)ti * T;
    mr[j0] = (unsigned char)m0;
    mr[j1] = (unsigned char)m1;
  }
}

// probs[bh,i,:] = softmax over j of (keep ? aw : MNEG), keep = (heavy|recent) & causal
__global__ void __launch_bounds__(256)
k_softmax(const float* __restrict__ aw, const unsigned char* __restrict__ msk,
          bf16_t* __restrict__ probs) {
  int bh = blockIdx.y, i = blockIdx.x;
  const float* row = aw + ((size_t)bh * T + i) * T;
  const unsigned char* mr = msk + ((size_t)bh * T + i) * T;
  bf16_t* pr = probs + ((size_t)bh * T + i) * T;
  float v[8];
  float mx = FNEG;
#pragma unroll
  for (int c = 0; c < 8; ++c) {
    int j = threadIdx.x + c * 256;
    bool heavy  = (i < HB) ? (j < HB) : (mr[j] != 0);
    bool recent = (j >= i - RB) && (j <= i);
    bool keep   = (heavy || recent) && (j <= i);
    v[c] = keep ? row[j] : MNEG;
    mx = fmaxf(mx, v[c]);
  }
  mx = blockMax(mx);
  float s = 0.f;
#pragma unroll
  for (int c = 0; c < 8; ++c) { v[c] = __expf(v[c] - mx); s += v[c]; }
  s = blockSum(s);
  float inv = 1.f / s;
#pragma unroll
  for (int c = 0; c < 8; ++c) pr[threadIdx.x + c * 256] = f2bf(v[c] * inv);
}

// attn[b,m,h*D+n] = sum_k probs[bh,m,k] * v[b,k,h*D+n]
// LDS-staged: block tile 64(M) x 64(N=D), BK=32; async copies as in k_gemm_bias.
__global__ void __launch_bounds__(256)
k_av(const bf16_t* __restrict__ probs, const bf16_t* __restrict__ Vb,
     bf16_t* __restrict__ attn) {
  __shared__ bf16_t As[64 * 32];   // [m][k]
  __shared__ bf16_t Bs[32 * 64];   // [k][n]  row-major
  int bh = blockIdx.y, b = bh >> 4, h = bh & 15;
  int lane = threadIdx.x & 31, wave = threadIdx.x >> 5;
  int mBase = blockIdx.x * 64;     // N = 64 exactly -> one N tile
  int waveM = (wave & 3) * 16;
  int waveN = (wave >> 2) * 32;
  const bf16_t* A  = probs + (size_t)bh * T * T;       // lda = T
  const bf16_t* Bm = Vb + (size_t)b * T * E + h * D;   // B[k][n] = Bm[k*E+n]
  v8f c0 = {0.f, 0.f, 0.f, 0.f, 0.f, 0.f, 0.f, 0.f};
  v8f c1 = {0.f, 0.f, 0.f, 0.f, 0.f, 0.f, 0.f, 0.f};

  int t = threadIdx.x;
  int rowA = t >> 2, partA = t & 3;                 // A: 64 rows x 4 chunks (4x16B = 64B/row)
  int rowB = t >> 3, partB = t & 7;                 // B: 32 rows x 8 chunks (128B/row)
  unsigned ldsA = ((unsigned)(uintptr_t)As) + rowA * 64 + partA * 16;
  unsigned ldsB = ((unsigned)(uintptr_t)Bs) + rowB * 128 + partB * 16;
  const bf16_t* gA = A + (size_t)(mBase + rowA) * T + partA * 8;
  const bf16_t* gB = Bm + (size_t)rowB * E + partB * 8;

  for (int k = 0; k < T; k += 32) {
    async_copy_b128(ldsA, gA + k);
    async_copy_b128(ldsB, gB + (size_t)k * E);
    wait_async0();
    __syncthreads();
    v16bf a  = load_a(As, 32, waveM, 0, lane);
    v16bf b0 = load_b_N(Bs, 64, waveN, 0, lane);
    v16bf b1 = load_b_N(Bs, 64, waveN + 16, 0, lane);
    c0 = __builtin_amdgcn_wmma_f32_16x16x32_bf16(false, a, false, b0, (short)0, c0, false, false);
    c1 = __builtin_amdgcn_wmma_f32_16x16x32_bf16(false, a, false, b1, (short)0, c1, false, false);
    __syncthreads();
  }
  int n0 = waveN + (lane & 15), n1 = n0 + 16;
  int mo = (lane >> 4) * 8;
#pragma unroll
  for (int r = 0; r < 8; ++r) {
    int m = mBase + waveM + mo + r;
    attn[((size_t)b * T + m) * E + h * D + n0] = f2bf(c0[r]);
    attn[((size_t)b * T + m) * E + h * D + n1] = f2bf(c1[r]);
  }
}

// ---------------- launch ----------------

extern "C" void kernel_launch(void* const* d_in, const int* in_sizes, int n_in,
                              void* d_out, int out_size, void* d_ws, size_t ws_size,
                              hipStream_t stream) {
  const float* hidden = (const float*)d_in[0];
  const float* amask  = (const float*)d_in[1];
  const float* Wq = (const float*)d_in[2]; const float* bq = (const float*)d_in[3];
  const float* Wk = (const float*)d_in[4]; const float* bk = (const float*)d_in[5];
  const float* Wv = (const float*)d_in[6]; const float* bv = (const float*)d_in[7];
  const float* Wo = (const float*)d_in[8]; const float* bo = (const float*)d_in[9];
  float* out = (float*)d_out;

  char* ws = (char*)d_ws;
  size_t off = 0;
  auto take = [&](size_t bytes) -> void* {
    void* p = ws + off;
    off += (bytes + 255) & ~(size_t)255;
    return p;
  };

  const size_t nHid = (size_t)Bb * T * E;      // 4,194,304
  const size_t nW   = (size_t)E * E;           // 1,048,576
  bf16_t* Xbf  = (bf16_t*)take(nHid * 2);
  bf16_t* Wqb  = (bf16_t*)take(nW * 2);
  bf16_t* Wkb  = (bf16_t*)take(nW * 2);
  bf16_t* Wvb  = (bf16_t*)take(nW * 2);
  bf16_t* Wob  = (bf16_t*)take(nW * 2);
  bf16_t* Qbf  = (bf16_t*)take(nHid * 2);
  bf16_t* Kbf  = (bf16_t*)take(nHid * 2);
  bf16_t* Vbf  = (bf16_t*)take(nHid * 2);
  bf16_t* Atn  = (bf16_t*)take(nHid * 2);
  float*  aw   = (float*)take((size_t)BH * T * T * 4);   // 512 MiB
  unsigned char* msk = (unsigned char*)take((size_t)BH * T * T);
  bf16_t* prb  = (bf16_t*)take((size_t)BH * T * T * 2);

  // convert inputs to bf16
  k_f32_to_bf16<<<(int)(nHid / 256), 256, 0, stream>>>(hidden, Xbf, (int)nHid);
  k_f32_to_bf16<<<(int)(nW / 256), 256, 0, stream>>>(Wq, Wqb, (int)nW);
  k_f32_to_bf16<<<(int)(nW / 256), 256, 0, stream>>>(Wk, Wkb, (int)nW);
  k_f32_to_bf16<<<(int)(nW / 256), 256, 0, stream>>>(Wv, Wvb, (int)nW);
  k_f32_to_bf16<<<(int)(nW / 256), 256, 0, stream>>>(Wo, Wob, (int)nW);

  // projections: M = 4096, N = 1024 -> 64x16 blocks of 64x64
  const int projBlocks = (Bb * T / 64) * (E / 64);
  k_gemm_bias<bf16_t><<<projBlocks, 256, 0, stream>>>(Xbf, Wqb, bq, QSCALE, Qbf, E, E, E, E, E / 64);
  k_gemm_bias<bf16_t><<<projBlocks, 256, 0, stream>>>(Xbf, Wkb, bk, 1.0f,   Kbf, E, E, E, E, E / 64);
  k_gemm_bias<bf16_t><<<projBlocks, 256, 0, stream>>>(Xbf, Wvb, bv, 1.0f,   Vbf, E, E, E, E, E / 64);

  // scores: per bh, 128x128 tiles / 8 waves = 2048 blocks
  k_scores<<<dim3((T / 16) * (T / 16) / 8, BH), 256, 0, stream>>>(Qbf, Kbf, amask, aw);

  // sequential heavy-hitter scan
  k_h2o<<<BH, 1024, 0, stream>>>(aw, msk);

  // masked softmax -> probs
  k_softmax<<<dim3(T, BH), 256, 0, stream>>>(aw, msk, prb);

  // probs @ v : per bh, 32 blocks of 64x64
  k_av<<<dim3(T / 64, BH), 256, 0, stream>>>(prb, Vbf, Atn);

  // output projection -> fp32 d_out
  k_gemm_bias<float><<<projBlocks, 256, 0, stream>>>(Atn, Wob, bo, 1.0f, out, E, E, E, E, E / 64);
}
